// ProMoELayer_74148315398462
// MI455X (gfx1250) — matrix operand order repaired and checked by reading, not compiled, not run on hardware
//
#include <hip/hip_runtime.h>
#include <hip/hip_bf16.h>
#include <math.h>

#define T_TOK 4096
#define DDIM  1024
#define HDIM  4096
#define NEXP  8
#define MT    32     // token tile (two 16-row WMMA subtiles)
#define MS    (MT/16)
#define HC    128    // h-chunk per fused stage
#define TILES_PER_E (T_TOK / MT)   // 128

typedef __attribute__((ext_vector_type(16))) __bf16 v16bf;
typedef __attribute__((ext_vector_type(8)))  __bf16 v8bf;
typedef __attribute__((ext_vector_type(8)))  float  v8f;

// ---- workspace layout (bytes) ----
#define WS_CNT 0                      // int cnt[8]
#define WS_SG  32                     // float Sg[8]
#define WS_SP  64                     // float Sp[8]
#define WS_TOK 128                    // int tokList[E*T]
#define WS_W   (128 + NEXP*T_TOK*4)   // float gateList[E*T]

// ---- pack two f32 -> packed bf16 dword ----
// HW cvt if the builtin exists; else round-half-away (2 adds + merge).
__device__ __forceinline__ unsigned pack_bf16(float f0, float f1) {
#if __has_builtin(__builtin_amdgcn_cvt_pk_bf16_f32)
    auto p = __builtin_amdgcn_cvt_pk_bf16_f32(f0, f1);
    unsigned u; __builtin_memcpy(&u, &p, 4); return u;
#else
    union { float f; unsigned u; } a, b; a.f = f0; b.f = f1;
    return ((a.u + 0x8000u) >> 16) | ((b.u + 0x8000u) & 0xffff0000u);
#endif
}
__device__ __forceinline__ __bf16 tobf(float f) {
    union { float f; unsigned u; } v; v.f = f;
    unsigned short h = (unsigned short)((v.u + 0x8000u) >> 16);
    __bf16 o; __builtin_memcpy(&o, &h, 2); return o;
}

// gelu_tanh(v) == v * sigmoid(1.5957691*(v + 0.044715 v^3)); branch-free
__device__ __forceinline__ float gelu_fast(float v) {
    float u2 = 1.5957691216057308f * (v + 0.044715f * v * v * v);
    float t  = __expf(-u2);
#if __has_builtin(__builtin_amdgcn_rcpf)
    return v * __builtin_amdgcn_rcpf(1.f + t);
#else
    return v / (1.f + t);
#endif
}

union AFrag { v16bf v; v8bf h[2]; };
union BFrag { v16bf v; unsigned u[8]; };

// ---------------- kernel 1: zero output + ws header ----------------
__global__ void k_zero(float* __restrict__ out, char* __restrict__ wsb) {
    size_t i = (size_t)blockIdx.x * blockDim.x + threadIdx.x;
    size_t stride = (size_t)gridDim.x * blockDim.x;
    size_t n = (size_t)T_TOK * DDIM + 1;
    for (size_t j = i; j < n; j += stride) out[j] = 0.f;
    if (i < 24) ((unsigned*)wsb)[i] = 0;
}

// ---------------- kernel 2: router (one wave per token) ----------------
__global__ __launch_bounds__(256) void k_router(const float* __restrict__ x,
                                                const float* __restrict__ wg,
                                                char* __restrict__ wsb) {
    __shared__ float sW[NEXP * DDIM];
    for (int i = threadIdx.x; i < NEXP * DDIM; i += 256) sW[i] = wg[i];
    __syncthreads();

    int wave = threadIdx.x >> 5, lane = threadIdx.x & 31;
    int t = blockIdx.x * 8 + wave;
    const float* xr = x + (size_t)t * DDIM;

    float acc[NEXP];
#pragma unroll
    for (int e = 0; e < NEXP; ++e) acc[e] = 0.f;
    for (int i = 0; i < DDIM / 32; ++i) {
        int d = lane + i * 32;
        float xv = xr[d];
#pragma unroll
        for (int e = 0; e < NEXP; ++e) acc[e] = fmaf(xv, sW[e * DDIM + d], acc[e]);
    }
#pragma unroll
    for (int e = 0; e < NEXP; ++e)
        for (int off = 16; off > 0; off >>= 1) acc[e] += __shfl_xor(acc[e], off, 32);

    if (lane == 0) {
        int i0 = 0; float v0 = acc[0];
#pragma unroll
        for (int e = 1; e < NEXP; ++e) if (acc[e] > v0) { v0 = acc[e]; i0 = e; }
        int i1 = (i0 == 0) ? 1 : 0; float v1 = acc[i1];
#pragma unroll
        for (int e = 0; e < NEXP; ++e) if (e != i0 && acc[e] > v1) { v1 = acc[e]; i1 = e; }
        float e1 = __expf(v1 - v0);
        float w0 = 1.f / (1.f + e1);
        float w1 = e1 * w0;
        float ps = 0.f, p[NEXP];
#pragma unroll
        for (int e = 0; e < NEXP; ++e) { p[e] = __expf(acc[e] - v0); ps += p[e]; }
        float inv = 1.f / ps;

        float* Sg = (float*)(wsb + WS_SG);
        float* Sp = (float*)(wsb + WS_SP);
#pragma unroll
        for (int e = 0; e < NEXP; ++e) unsafeAtomicAdd(&Sp[e], p[e] * inv);
        unsafeAtomicAdd(&Sg[i0], w0);
        unsafeAtomicAdd(&Sg[i1], w1);

        int*   cnt = (int*)(wsb + WS_CNT);
        int*   tok = (int*)(wsb + WS_TOK);
        float* gl  = (float*)(wsb + WS_W);
        int p0 = atomicAdd(&cnt[i0], 1); tok[i0 * T_TOK + p0] = t; gl[i0 * T_TOK + p0] = w0;
        int p1 = atomicAdd(&cnt[i1], 1); tok[i1 * T_TOK + p1] = t; gl[i1 * T_TOK + p1] = w1;
    }
}

// ---------------- kernel 3: fused expert FFN (WMMA bf16) ----------------
__global__ __launch_bounds__(256) void k_ffn(const float* __restrict__ x,
                                             const float* __restrict__ W1,
                                             const float* __restrict__ b1,
                                             const float* __restrict__ W2,
                                             const float* __restrict__ b2,
                                             const char* __restrict__ wsb,
                                             float* __restrict__ y) {
    __shared__ __bf16 Xb[MT * DDIM];   // 64 KB
    __shared__ __bf16 Hb[MT * HC];     // 8 KB
    __shared__ int    tokSh[MT];
    __shared__ float  gateSh[MT];
    __shared__ int    cntSh;

    int e    = blockIdx.x >> 7;              // expert-major for L2 reuse
    int tile = blockIdx.x & (TILES_PER_E - 1);

    if (threadIdx.x == 0) cntSh = ((const int*)(wsb + WS_CNT))[e];
    __syncthreads();
    int n_e = cntSh;
    int base = tile * MT;
    if (base >= n_e) return;

    const int*   tok = (const int*)(wsb + WS_TOK) + e * T_TOK;
    const float* gl  = (const float*)(wsb + WS_W) + e * T_TOK;
    if (threadIdx.x < MT) {
        int r = threadIdx.x, src = base + r;
        bool valid = src < n_e;
        tokSh[r]  = tok[valid ? src : base];
        gateSh[r] = valid ? gl[src] : 0.f;
    }
    __syncthreads();
    {
        unsigned* Xu = (unsigned*)Xb;
        for (int idx = threadIdx.x; idx < MT * (DDIM / 2); idx += 256) {
            int r = idx >> 9;
            int c2 = idx & 511;
            float2 f = *(const float2*)&x[(size_t)tokSh[r] * DDIM + 2 * c2];
            Xu[idx] = pack_bf16(f.x, f.y);
        }
    }
    __syncthreads();

    const float* W1e = W1 + (size_t)e * DDIM * HDIM;
    const float* W2e = W2 + (size_t)e * HDIM * DDIM;
    const float* b1e = b1 + e * HDIM;
    const float* b2e = b2 + e * DDIM;

    int wave = threadIdx.x >> 5, lane = threadIdx.x & 31;
    int lmod = lane & 15, lhi = lane >> 4;

    v8f acc2[8][MS];
#pragma unroll
    for (int i = 0; i < 8; ++i)
#pragma unroll
        for (int m = 0; m < MS; ++m)
#pragma unroll
            for (int j = 0; j < 8; ++j) acc2[i][m][j] = 0.f;

    for (int hc = 0; hc < HDIM / HC; ++hc) {
        // ---- GEMM1: wave computes h-cols [hb,hb+16) for MS subtiles ----
        int hb   = hc * HC + wave * 16;
        int hcol = hb + lmod;
        v8f c1[MS];
#pragma unroll
        for (int m = 0; m < MS; ++m)
#pragma unroll
            for (int j = 0; j < 8; ++j) c1[m][j] = 0.f;

        // raw-B gather (16 strided f32), register double-buffered across K steps
        auto loadW1 = [&](int kk, float* dst) {
            const float* wp = &W1e[(size_t)(kk * 32 + lhi * 16) * HDIM + hcol];
            __builtin_prefetch(wp + 64 * HDIM, 0, 0);
#pragma unroll
            for (int i = 0; i < 16; ++i) dst[i] = wp[i * HDIM];
        };
        auto wmma1 = [&](const float* braw, int kk) {
            BFrag b;
#pragma unroll
            for (int i = 0; i < 8; ++i) b.u[i] = pack_bf16(braw[2 * i], braw[2 * i + 1]);
            int k0 = kk * 32;
#pragma unroll
            for (int m = 0; m < MS; ++m) {
                AFrag a;
                a.h[0] = *(const v8bf*)&Xb[(m * 16 + lmod) * DDIM + k0 + lhi * 16];
                a.h[1] = *(const v8bf*)&Xb[(m * 16 + lmod) * DDIM + k0 + lhi * 16 + 8];
                c1[m] = __builtin_amdgcn_wmma_f32_16x16x32_bf16(false, a.v, false, b.v,
                                                                (short)0, c1[m], false, false);
            }
        };
        {
            float bufA[16], bufB[16];
            loadW1(0, bufA);
            for (int kk = 0; kk < DDIM / 32; kk += 2) {
                loadW1(kk + 1, bufB);
                wmma1(bufA, kk);
                if (kk + 2 < DDIM / 32) loadW1(kk + 2, bufA);
                wmma1(bufB, kk + 1);
            }
        }
        // bias + gelu -> Hb
        float bias = b1e[hcol];
        __syncthreads();
#pragma unroll
        for (int m = 0; m < MS; ++m)
#pragma unroll
            for (int r = 0; r < 8; ++r) {
                int row = m * 16 + r + 8 * lhi;
                Hb[row * HC + wave * 16 + lmod] = tobf(gelu_fast(c1[m][r] + bias));
            }
        __syncthreads();

        // ---- GEMM2 partial: K = this h-chunk, wave's 128 d-cols ----
#pragma unroll 2
        for (int kk = 0; kk < HC / 32; ++kk) {
            int k0 = kk * 32;
            AFrag a[MS];
#pragma unroll
            for (int m = 0; m < MS; ++m) {
                a[m].h[0] = *(const v8bf*)&Hb[(m * 16 + lmod) * HC + k0 + lhi * 16];
                a[m].h[1] = *(const v8bf*)&Hb[(m * 16 + lmod) * HC + k0 + lhi * 16 + 8];
            }
            int kglob = hc * HC + k0 + lhi * 16;
#pragma unroll
            for (int nf = 0; nf < 8; ++nf) {
                int dcol = wave * 128 + nf * 16 + lmod;
                const float* wp = &W2e[(size_t)kglob * DDIM + dcol];
                __builtin_prefetch(wp + 32 * DDIM, 0, 0);
                BFrag b;
#pragma unroll
                for (int i = 0; i < 8; ++i)
                    b.u[i] = pack_bf16(wp[(2 * i) * DDIM], wp[(2 * i + 1) * DDIM]);
#pragma unroll
                for (int m = 0; m < MS; ++m)
                    acc2[nf][m] = __builtin_amdgcn_wmma_f32_16x16x32_bf16(false, a[m].v, false, b.v,
                                                                          (short)0, acc2[nf][m],
                                                                          false, false);
            }
        }
    }

    // ---- epilogue: +b2, gate-scale, atomic accumulate ----
#pragma unroll
    for (int nf = 0; nf < 8; ++nf) {
        int dcol = wave * 128 + nf * 16 + lmod;
        float bias2 = b2e[dcol];
#pragma unroll
        for (int m = 0; m < MS; ++m)
#pragma unroll
            for (int r = 0; r < 8; ++r) {
                int row = m * 16 + r + 8 * lhi;
                float v = acc2[nf][m][r] + bias2;
                unsafeAtomicAdd(&y[(size_t)tokSh[row] * DDIM + dcol], gateSh[row] * v);
            }
    }
}

// ---------------- kernel 4: aux loss ----------------
__global__ void k_aux(const char* __restrict__ wsb, float* __restrict__ out) {
    if (threadIdx.x == 0 && blockIdx.x == 0) {
        const float* Sg = (const float*)(wsb + WS_SG);
        const float* Sp = (const float*)(wsb + WS_SP);
        float s = 0.f;
        const float invT = 1.f / (float)T_TOK;
#pragma unroll
        for (int e = 0; e < NEXP; ++e) s += (Sg[e] * invT) * (Sp[e] * invT);
        out[(size_t)T_TOK * DDIM] = (float)NEXP * s;
    }
}

extern "C" void kernel_launch(void* const* d_in, const int* in_sizes, int n_in,
                              void* d_out, int out_size, void* d_ws, size_t ws_size,
                              hipStream_t stream) {
    const float* x  = (const float*)d_in[0];
    const float* wg = (const float*)d_in[1];
    const float* W1 = (const float*)d_in[2];
    const float* b1 = (const float*)d_in[3];
    const float* W2 = (const float*)d_in[4];
    const float* b2 = (const float*)d_in[5];
    float* out = (float*)d_out;
    char*  wsb = (char*)d_ws;

    k_zero<<<dim3(2048), dim3(256), 0, stream>>>(out, wsb);
    k_router<<<dim3(T_TOK / 8), dim3(256), 0, stream>>>(x, wg, wsb);
    k_ffn<<<dim3(NEXP * TILES_PER_E), dim3(256), 0, stream>>>(x, W1, b1, W2, b2, wsb, out);
    k_aux<<<dim3(1), dim3(32), 0, stream>>>(wsb, out);
}
